// MessageGeneratorRNN_70918499991976
// MI455X (gfx1250) — matrix operand order repaired and checked by reading, not compiled, not run on hardware
//
#include <hip/hip_runtime.h>

// ---------------------------------------------------------------------------
// Gumbel-softmax RNN generator for MI455X (gfx1250).
// 16 rows per workgroup, all 12 scan steps executed locally (rows are
// independent across the scan). bf16 WMMA (V_WMMA_F32_16X16X32_BF16) for all
// four GEMMs with f32 accumulation; softmax in f32. Weights pre-swizzled to
// the bf16 WMMA-B lane layout in d_ws so each lane fetches its operand with
// two coalesced b128 loads at an SGPR base + lane*32 + immediate offset.
// A zero SGPR offset is laundered per step (not the pointers!) so weight
// loads stay step-variant (no LICM->scratch spills) while keeping global
// address-space provenance (global_load + saddr, not flat_load).
// ---------------------------------------------------------------------------

typedef __attribute__((ext_vector_type(16))) __bf16 v16bf;
typedef __attribute__((ext_vector_type(8)))  float  v8f;

#define N_ROWS 4096
#define HID    1024
#define VOCAB  1024
#define EMB    256
#define NOS    12

#define PITCH_H 1032        // halves per LDS row for 1024-wide activations
#define PITCH_E 264         // halves per LDS row for 256-wide activations
#define PU_H (PITCH_H / 2)  // uint pitch
#define PU_E (PITCH_E / 2)

__device__ __forceinline__ unsigned short f2bf(float f) {
  unsigned u = __float_as_uint(f);
  u += 0x7fffu + ((u >> 16) & 1u);   // round-to-nearest-even
  return (unsigned short)(u >> 16);
}

union ABf { v16bf v; unsigned int u[8]; uint4 q[2]; };

// A operand (16x32 bf16, M x K) from LDS, dense-A layout (ISA 7.12.2):
//   lane m = lane&15, hi = lane>>4; VGPR v pair at
//   k = kt*32 + (v<4?0:16) + hi*8 + 2*(v&3)  -> uint offset kt*16 + (v<4?0:8) + hi*4 + (v&3)
// Consecutive uints 0..3 and 8..11 -> lowers to two ds_load_b128.
__device__ __forceinline__ v16bf load_a(const unsigned int* lds, int m, int hi,
                                        int kt, int pitch_u) {
  ABf r;
  const int base = m * pitch_u + kt * 16 + hi * 4;
#pragma unroll
  for (int v = 0; v < 8; ++v)
    r.u[v] = lds[base + ((v < 4) ? 0 : 8) + (v & 3)];
  return r.v;
}

// One K-slice (kt) of an 8-N-tile GEMM: load all 8 B tiles + the A tile,
// then run the 8-WMMA chain. tile_base is wave-uniform (scalar) so B
// addresses become saddr + lane*32 + imm(t*1024).
__device__ __forceinline__ void gemm8_step(const unsigned int* lds, int pitch_u,
                                           const unsigned short* __restrict__ Wsw,
                                           int tile_base, int l16, int hi,
                                           int lane, int kt, v8f acc[8]) {
  ABf bs[8];
  const uint4* bp = (const uint4*)Wsw + ((size_t)tile_base * 32 + lane) * 2;
#pragma unroll
  for (int t = 0; t < 8; ++t) {
    bs[t].q[0] = bp[t * 64];
    bs[t].q[1] = bp[t * 64 + 1];
  }
  v16bf a = load_a(lds, l16, hi, kt, pitch_u);
#pragma unroll
  for (int t = 0; t < 8; ++t)
    acc[t] = __builtin_amdgcn_wmma_f32_16x16x32_bf16(
        false, a, false, bs[t].v, (short)0, acc[t], false, false);
}

// ---------------------------------------------------------------------------
// Prep: swizzle a row-major f32 weight W[Nw][Kw] (B[k][n] = W[n][k]) into the
// bf16 per-lane WMMA-B order. One output half per thread.
//   B layout (ISA 7.12.4 pattern): n = ntile*16 + (lane&15),
//   k = ktile*32 + (v<4 ? 0 : 16) + (lane>=16 ? 8 : 0) + 2*(v&3) + p,  half index in lane = 2v+p.
// ---------------------------------------------------------------------------
__global__ void swizzleB(const float* __restrict__ W, unsigned short* __restrict__ out,
                         int Kw, int NT, int total) {
  int idx = blockIdx.x * blockDim.x + threadIdx.x;
  if (idx >= total) return;
  int h     = idx & 15;
  int v     = h >> 1;
  int p     = h & 1;
  int lane  = (idx >> 4) & 31;
  int tmp   = idx >> 9;
  int ntile = tmp % NT;
  int ktile = tmp / NT;
  int n = ntile * 16 + (lane & 15);
  int k = ktile * 32 + ((v < 4) ? 0 : 16) + ((lane >> 4) ? 8 : 0) + 2 * (v & 3) + p;
  out[idx] = f2bf(W[(size_t)n * Kw + k]);
}

// ---------------------------------------------------------------------------
// Main kernel: one 16-row tile per block, 8 waves (256 threads), 12 steps.
// Wave w handles column tiles [w*8, w*8+8) of 64 for HID/VOCAB outputs and
// tiles [w*2, w*2+2) of 16 for EMB outputs.
// ---------------------------------------------------------------------------
__global__ __launch_bounds__(256) void rnn_step_kernel(
    const float* __restrict__ target, const float* __restrict__ gum,
    const float* __restrict__ sos,
    const float* __restrict__ b_ih, const float* __restrict__ b_hh,
    const float* __restrict__ b_out, const float* __restrict__ b_emb,
    const unsigned short* __restrict__ Wih_sw,
    const unsigned short* __restrict__ Whh_sw,
    const unsigned short* __restrict__ Wout_sw,
    const unsigned short* __restrict__ Wemb_sw,
    float* __restrict__ out)
{
  __shared__ unsigned int h_u[16 * PU_H];   // h   (bf16) 33 KB
  __shared__ unsigned int x_u[16 * PU_H];   // x   (bf16) 33 KB
  __shared__ unsigned int e_u[16 * PU_E];   // e   (bf16)  8 KB
  __shared__ float red_p[8 * 16];           // per-wave row partials
  __shared__ float red_f[16];               // final row max / 1/sum

  const int tid  = threadIdx.x;
  // wave id is uniform across the wave -> force scalar so all weight-tile
  // indices become SGPR math (saddr addressing, no per-tile v_add_nc_u64).
  const int wave = __builtin_amdgcn_readfirstlane(tid >> 5);
  const int lane = tid & 31;
  const int hi   = (lane >> 4) & 1;
  const int l16  = lane & 15;
  const int base = blockIdx.x * 16;

  unsigned short* hs = (unsigned short*)h_u;
  unsigned short* xs = (unsigned short*)x_u;
  unsigned short* es = (unsigned short*)e_u;

  // init: h = target rows, e = sos broadcast
  for (int i = tid; i < 16 * HID; i += 256) {
    int m = i >> 10, k = i & (HID - 1);
    hs[m * PITCH_H + k] = f2bf(target[(size_t)(base + m) * HID + k]);
  }
  for (int i = tid; i < 16 * EMB; i += 256) {
    int m = i >> 8, k = i & (EMB - 1);
    es[m * PITCH_E + k] = f2bf(sos[k]);
  }
  __syncthreads();

  const v8f vzero = {0.f, 0.f, 0.f, 0.f, 0.f, 0.f, 0.f, 0.f};
  const int ncb = wave * 8;   // first of this wave's 8 column tiles (of 64)

  // Step-invariant biases: hoist into registers (18 scalars / lane).
  float bias_h[8], bo_r[8], be_r[2];
#pragma unroll
  for (int t = 0; t < 8; ++t) {
    int n = (ncb + t) * 16 + l16;
    bias_h[t] = b_ih[n] + b_hh[n];
    bo_r[t]   = b_out[n];
  }
#pragma unroll
  for (int t = 0; t < 2; ++t) be_r[t] = b_emb[(wave * 2 + t) * 16 + l16];

  // Opaque zero offset (SGPR): re-laundered each step so weight addresses
  // are not loop-invariant (prevents hoist+spill), while the pointers keep
  // their kernel-arg provenance (global address space, saddr addressing).
  size_t zoff = 0;

  for (int step = 0; step < NOS; ++step) {
    asm volatile("" : "+s"(zoff));
    const unsigned short* wih  = Wih_sw  + zoff;
    const unsigned short* whh  = Whh_sw  + zoff;
    const unsigned short* wout = Wout_sw + zoff;
    const unsigned short* wemb = Wemb_sw + zoff;

    // ---------------- GEMM1: hpre = e @ W_ih^T + h @ W_hh^T ----------------
    v8f acc[8];
#pragma unroll
    for (int t = 0; t < 8; ++t) acc[t] = vzero;

    for (int kt = 0; kt < EMB / 32; ++kt)
      gemm8_step(e_u, PU_E, wih, kt * 64 + ncb, l16, hi, lane, kt, acc);
    for (int kt = 0; kt < HID / 32; ++kt)
      gemm8_step(h_u, PU_H, whh, kt * 64 + ncb, l16, hi, lane, kt, acc);

    // h = tanh(hpre + b_ih + b_hh)
#pragma unroll
    for (int t = 0; t < 8; ++t)
#pragma unroll
      for (int v = 0; v < 8; ++v) acc[t][v] = tanhf(acc[t][v] + bias_h[t]);
    __syncthreads();   // all GEMM1 reads of h_u / e_u complete
#pragma unroll
    for (int t = 0; t < 8; ++t) {
      int n = (ncb + t) * 16 + l16;
#pragma unroll
      for (int v = 0; v < 8; ++v) hs[(v + hi * 8) * PITCH_H + n] = f2bf(acc[t][v]);
    }
    __syncthreads();   // new h visible

    // ---------------- GEMM2: logits = h @ W_out^T --------------------------
#pragma unroll
    for (int t = 0; t < 8; ++t) acc[t] = vzero;
    for (int kt = 0; kt < HID / 32; ++kt)
      gemm8_step(h_u, PU_H, wout, kt * 64 + ncb, l16, hi, lane, kt, acc);

    // + b_out + gumbel noise
#pragma unroll
    for (int t = 0; t < 8; ++t) {
      int n = (ncb + t) * 16 + l16;
#pragma unroll
      for (int v = 0; v < 8; ++v) {
        int m = v + hi * 8;
        acc[t][v] += bo_r[t] + gum[((size_t)step * N_ROWS + base + m) * VOCAB + n];
      }
    }

    // ---------------- softmax over 1024 columns per row --------------------
    float pm[8];
#pragma unroll
    for (int v = 0; v < 8; ++v) {
      float x = acc[0][v];
#pragma unroll
      for (int t = 1; t < 8; ++t) x = fmaxf(x, acc[t][v]);
      pm[v] = x;
    }
#pragma unroll
    for (int off = 1; off <= 8; off <<= 1)
#pragma unroll
      for (int v = 0; v < 8; ++v) pm[v] = fmaxf(pm[v], __shfl_xor(pm[v], off, 32));
    if (l16 == 0) {
#pragma unroll
      for (int v = 0; v < 8; ++v) red_p[wave * 16 + hi * 8 + v] = pm[v];
    }
    __syncthreads();
    if (tid < 16) {
      float mm = red_p[tid];
      for (int w = 1; w < 8; ++w) mm = fmaxf(mm, red_p[w * 16 + tid]);
      red_f[tid] = mm;
    }
    __syncthreads();

    float rmv[8], ps[8];
#pragma unroll
    for (int v = 0; v < 8; ++v) { rmv[v] = red_f[hi * 8 + v]; ps[v] = 0.0f; }
#pragma unroll
    for (int t = 0; t < 8; ++t)
#pragma unroll
      for (int v = 0; v < 8; ++v) {
        float ex = __expf(acc[t][v] - rmv[v]);
        acc[t][v] = ex;
        ps[v] += ex;
      }
#pragma unroll
    for (int off = 1; off <= 8; off <<= 1)
#pragma unroll
      for (int v = 0; v < 8; ++v) ps[v] += __shfl_xor(ps[v], off, 32);
    if (l16 == 0) {
#pragma unroll
      for (int v = 0; v < 8; ++v) red_p[wave * 16 + hi * 8 + v] = ps[v];
    }
    __syncthreads();
    if (tid < 16) {
      float s = red_p[tid];
      for (int w = 1; w < 8; ++w) s += red_p[w * 16 + tid];
      red_f[tid] = 1.0f / s;
    }
    __syncthreads();

    // scale; write x to global output [N][NOS][VOCAB] and to LDS as bf16
#pragma unroll
    for (int v = 0; v < 8; ++v) rmv[v] = red_f[hi * 8 + v];   // now 1/sum
#pragma unroll
    for (int t = 0; t < 8; ++t) {
      int n = (ncb + t) * 16 + l16;
#pragma unroll
      for (int v = 0; v < 8; ++v) {
        int m = v + hi * 8;
        float x = acc[t][v] * rmv[v];
        out[((size_t)(base + m) * NOS + step) * VOCAB + n] = x;
        xs[m * PITCH_H + n] = f2bf(x);
      }
    }
    __syncthreads();   // x visible

    // ---------------- GEMM3: e_next = x @ W_emb^T --------------------------
    v8f acc3[2];
    acc3[0] = vzero; acc3[1] = vzero;
    for (int kt = 0; kt < VOCAB / 32; ++kt) {
      ABf bs[2];
      const uint4* bp = (const uint4*)wemb +
                        ((size_t)(kt * 16 + wave * 2) * 32 + lane) * 2;
      bs[0].q[0] = bp[0];  bs[0].q[1] = bp[1];
      bs[1].q[0] = bp[64]; bs[1].q[1] = bp[65];
      v16bf a = load_a(x_u, l16, hi, kt, PU_H);
      acc3[0] = __builtin_amdgcn_wmma_f32_16x16x32_bf16(
          false, a, false, bs[0].v, (short)0, acc3[0], false, false);
      acc3[1] = __builtin_amdgcn_wmma_f32_16x16x32_bf16(
          false, a, false, bs[1].v, (short)0, acc3[1], false, false);
    }
#pragma unroll
    for (int t = 0; t < 2; ++t) {
      int n = (wave * 2 + t) * 16 + l16;
#pragma unroll
      for (int v = 0; v < 8; ++v)
        es[(v + hi * 8) * PITCH_E + n] = f2bf(acc3[t][v] + be_r[t]);
    }
    __syncthreads();   // e_next visible for next step
  }
}

// ---------------------------------------------------------------------------
// Launch. d_ws layout (bf16 halves):
//   [0)         W_ih  swizzled:  8*64*512  =  262144
//   [262144)    W_hh  swizzled: 32*64*512  = 1048576
//   [1310720)   W_out swizzled: 32*64*512  = 1048576
//   [2359296)   W_emb swizzled: 32*16*512  =  262144
//   total 2621440 halves = 5.0 MiB
// ---------------------------------------------------------------------------
extern "C" void kernel_launch(void* const* d_in, const int* in_sizes, int n_in,
                              void* d_out, int out_size, void* d_ws, size_t ws_size,
                              hipStream_t stream) {
  const float* target = (const float*)d_in[0];
  const float* gum    = (const float*)d_in[1];
  const float* sos    = (const float*)d_in[2];
  const float* W_ih   = (const float*)d_in[3];
  const float* b_ih   = (const float*)d_in[4];
  const float* W_hh   = (const float*)d_in[5];
  const float* b_hh   = (const float*)d_in[6];
  const float* W_out  = (const float*)d_in[7];
  const float* b_out  = (const float*)d_in[8];
  const float* W_emb  = (const float*)d_in[9];
  const float* b_emb  = (const float*)d_in[10];

  unsigned short* ws      = (unsigned short*)d_ws;
  unsigned short* Wih_sw  = ws;
  unsigned short* Whh_sw  = ws + 262144;
  unsigned short* Wout_sw = ws + 262144 + 1048576;
  unsigned short* Wemb_sw = ws + 262144 + 1048576 + 1048576;

  swizzleB<<<262144 / 256, 256, 0, stream>>>(W_ih,  Wih_sw,  256,  64, 262144);
  swizzleB<<<1048576 / 256, 256, 0, stream>>>(W_hh,  Whh_sw,  1024, 64, 1048576);
  swizzleB<<<1048576 / 256, 256, 0, stream>>>(W_out, Wout_sw, 1024, 64, 1048576);
  swizzleB<<<262144 / 256, 256, 0, stream>>>(W_emb, Wemb_sw, 1024, 16, 262144);

  rnn_step_kernel<<<N_ROWS / 16, 256, 0, stream>>>(
      target, gum, sos, b_ih, b_hh, b_out, b_emb,
      Wih_sw, Whh_sw, Wout_sw, Wemb_sw, (float*)d_out);

  (void)in_sizes; (void)n_in; (void)out_size; (void)ws_size;
}
